// PQMatcher_25744033973069
// MI455X (gfx1250) — matrix operand order repaired and checked by reading, not compiled, not run on hardware
//
#include <hip/hip_runtime.h>
#include <hip/hip_bf16.h>
#include <stdint.h>

#define B_  128
#define LP_ 256
#define LQ_ 256
#define IN_ 768
#define H_  768

typedef __attribute__((ext_vector_type(16))) __bf16 v16bf;
typedef __attribute__((ext_vector_type(8)))  float  v8f;

// ---------- helpers ----------
static __device__ __forceinline__ unsigned short f32_to_bf16_bits(float f) {
  union { float f; uint32_t u; } x; x.f = f;
  uint32_t u = x.u;
  uint32_t r = (u + 0x7FFFu + ((u >> 16) & 1u)) >> 16; // RNE
  if ((u & 0x7F800000u) == 0x7F800000u && (u & 0x007FFFFFu)) r = (u >> 16) | 1u; // keep NaN
  return (unsigned short)r;
}
static __device__ __forceinline__ float bf16_bits_to_f32(unsigned short s) {
  union { uint32_t u; float f; } x; x.u = ((uint32_t)s) << 16; return x.f;
}
static __device__ __forceinline__ float sigmoidf_(float x) {
  return 1.0f / (1.0f + __expf(-x));
}

// ---------- elementwise / prep kernels ----------
__global__ void convert_to_bf16_kernel(const float* __restrict__ src,
                                       unsigned short* __restrict__ dst, long n) {
  long i = (long)blockIdx.x * blockDim.x + threadIdx.x;
  if (i < n) dst[i] = f32_to_bf16_bits(src[i]);
}

__global__ void transpose_to_bf16_kernel(const float* __restrict__ src,
                                         unsigned short* __restrict__ dst,
                                         int rows, int cols) {
  long i = (long)blockIdx.x * blockDim.x + threadIdx.x;
  long n = (long)rows * cols;
  if (i >= n) return;
  int r = (int)(i / cols);
  int c = (int)(i % cols);
  dst[(long)c * rows + r] = f32_to_bf16_bits(src[i]);
}

__global__ void init_v_kernel(float* __restrict__ v, unsigned short* __restrict__ vbf, int n) {
  int i = blockIdx.x * blockDim.x + threadIdx.x;
  if (i < n) { v[i] = 0.0f; vbf[i] = 0; }
}

// ---------- WMMA bf16 GEMM: C[MxN] = A[MxK] * B[KxN] (+ bias) ----------
// A: bf16 row-major MxK.  BT: bf16 row-major NxK (i.e. B transposed).
// One wave computes a 16x64 strip (4 WMMA tiles reusing the A fragment).
// bias (optional, f32): C[m][n] += bias[m*biasRowStride + biasOffset + n]
//   (biasRowStride=0 -> vector bias of length N).
__global__ void wmma_gemm_bf16_kernel(
    const unsigned short* __restrict__ A,
    const unsigned short* __restrict__ BT,
    const float* __restrict__ bias, long biasRowStride, long biasOffset,
    float* __restrict__ Cf, unsigned short* __restrict__ Cb,
    int M, int N, int K)
{
  const int lane = threadIdx.x & 31;
  const int wave = blockIdx.x * (blockDim.x >> 5) + (threadIdx.x >> 5);
  const int stripsN = N >> 6;
  const int strips  = (M >> 4) * stripsN;
  if (wave >= strips) return;
  const int tm = (wave / stripsN) << 4;
  const int tn = (wave % stripsN) << 6;

  const int mrow = lane & 15;   // A row within tile / BT row within tile
  const int half = lane >> 4;   // K-half selector: half h holds k = 16h..16h+15 (contiguous order)

  const unsigned short* Arow  = A  + (long)(tm + mrow) * K + (half << 4);
  const unsigned short* Brow0 = BT + (long)(tn + mrow) * K + (half << 4);
  const long bstep = (long)16 * K;

  v8f acc0 = {}, acc1 = {}, acc2 = {}, acc3 = {};
  union Frag { v16bf v; uint4 u[2]; };

  for (int k0 = 0; k0 < K; k0 += 32) {
    __builtin_prefetch((const void*)(Arow + k0 + 64), 0, 3);   // global_prefetch_b8 (near)
    Frag a;
    a.u[0] = *(const uint4*)(Arow + k0);
    a.u[1] = *(const uint4*)(Arow + k0 + 8);
    Frag b0, b1, b2, b3;
    b0.u[0] = *(const uint4*)(Brow0 + k0);
    b0.u[1] = *(const uint4*)(Brow0 + k0 + 8);
    b1.u[0] = *(const uint4*)(Brow0 + bstep + k0);
    b1.u[1] = *(const uint4*)(Brow0 + bstep + k0 + 8);
    b2.u[0] = *(const uint4*)(Brow0 + 2 * bstep + k0);
    b2.u[1] = *(const uint4*)(Brow0 + 2 * bstep + k0 + 8);
    b3.u[0] = *(const uint4*)(Brow0 + 3 * bstep + k0);
    b3.u[1] = *(const uint4*)(Brow0 + 3 * bstep + k0 + 8);
    acc0 = __builtin_amdgcn_wmma_f32_16x16x32_bf16(false, a.v, false, b0.v, (short)0, acc0, false, false);
    acc1 = __builtin_amdgcn_wmma_f32_16x16x32_bf16(false, a.v, false, b1.v, (short)0, acc1, false, false);
    acc2 = __builtin_amdgcn_wmma_f32_16x16x32_bf16(false, a.v, false, b2.v, (short)0, acc2, false, false);
    acc3 = __builtin_amdgcn_wmma_f32_16x16x32_bf16(false, a.v, false, b3.v, (short)0, acc3, false, false);
  }

  const int ncol = lane & 15;
  v8f accs[4] = { acc0, acc1, acc2, acc3 };
  for (int j = 0; j < 4; ++j) {
    v8f acc = accs[j];
    const int cn = tn + (j << 4) + ncol;
    for (int r = 0; r < 8; ++r) {
      const int cm = tm + r + (half << 3);
      float val = acc[r];
      if (bias) val += bias[(long)cm * biasRowStride + biasOffset + cn];
      const long idx = (long)cm * N + cn;
      if (Cf) Cf[idx] = val;
      if (Cb) Cb[idx] = f32_to_bf16_bits(val);
    }
  }
}

// ---------- per-step kernels ----------
// s[b][l] = sum_h tanh(WuqUq_bf16[b][l][h] + bias[b][h]) * V[h]
// The 8 rows this block consumes (12 KB bf16) are bulk-DMA'd into LDS with
// global_load_async_to_lds_b128 (ASYNCcnt), then each wave reduces one row.
__global__ __launch_bounds__(256) void attn_score_kernel(
    const unsigned short* __restrict__ WuqUq,
    const float* __restrict__ bias,
    const float* __restrict__ V,
    float* __restrict__ s)
{
  __shared__ unsigned short tile[8 * H_];   // 12 KB
  __shared__ float sb[H_];                  // bias row
  __shared__ float sv[H_];                  // V
  const int b  = blockIdx.x / (LQ_ / 8);
  const int l0 = (blockIdx.x % (LQ_ / 8)) * 8;
  const int tid = threadIdx.x;

  // async DMA: 8*768 bf16 = 768 x 16-byte chunks, 3 per thread
  {
    const unsigned long long src =
        (unsigned long long)(uintptr_t)(WuqUq + ((long)b * LQ_ + l0) * H_);
    const unsigned ldsBase = (unsigned)(uintptr_t)(&tile[0]); // low 32 bits = LDS offset
    #pragma unroll
    for (int c = tid; c < 768; c += 256) {
      unsigned       ldsa = ldsBase + (unsigned)c * 16u;
      unsigned long long g = src + (unsigned long long)c * 16ull;
      asm volatile("global_load_async_to_lds_b128 %0, %1, off"
                   :: "v"(ldsa), "v"(g) : "memory");
    }
  }
  // stage bias + V while the DMA is in flight
  for (int i = tid; i < H_; i += 256) {
    sb[i] = bias[(long)b * H_ + i];
    sv[i] = V[i];
  }
  asm volatile("s_wait_asynccnt 0" ::: "memory");
  __syncthreads();

  const int wave = tid >> 5;
  const int lane = tid & 31;
  const unsigned short* xr = &tile[wave * H_];
  float acc = 0.0f;
  #pragma unroll
  for (int j = 0; j < 3; ++j) {
    const int base = (lane + 32 * j) * 8;
    union { uint4 u; unsigned short h[8]; } p;
    p.u = *(const uint4*)(xr + base);            // ds_load_b128
    float bb[8], vv[8];
    *(float4*)&bb[0] = *(const float4*)(sb + base);
    *(float4*)&bb[4] = *(const float4*)(sb + base + 4);
    *(float4*)&vv[0] = *(const float4*)(sv + base);
    *(float4*)&vv[4] = *(const float4*)(sv + base + 4);
    #pragma unroll
    for (int e = 0; e < 8; ++e)
      acc += tanhf(bf16_bits_to_f32(p.h[e]) + bb[e]) * vv[e];
  }
  for (int off = 16; off > 0; off >>= 1) acc += __shfl_down(acc, off);
  if (lane == 0) s[(long)b * LQ_ + (l0 + wave)] = acc;
}

__global__ void softmax_kernel(const float* __restrict__ s, float* __restrict__ a) {
  const int b = blockIdx.x;
  const int l = threadIdx.x;          // LQ_ == blockDim == 256
  __shared__ float red[LQ_];
  float val = s[(long)b * LQ_ + l];
  red[l] = val; __syncthreads();
  for (int off = 128; off > 0; off >>= 1) {
    if (l < off) red[l] = fmaxf(red[l], red[l + off]);
    __syncthreads();
  }
  float mx = red[0]; __syncthreads();
  float e = __expf(val - mx);
  red[l] = e; __syncthreads();
  for (int off = 128; off > 0; off >>= 1) {
    if (l < off) red[l] += red[l + off];
    __syncthreads();
  }
  a[(long)b * LQ_ + l] = e / red[0];
}

// c[b][f] = sum_l a[b][l]*Uq[b][l][f];   r = [Up[b,t,:], c]; also bf16 copy of r
// 96 threads/block; each thread owns one 8-wide bf16 chunk -> one b128 per row.
__global__ __launch_bounds__(96) void context_kernel(
    const float* __restrict__ a,
    const unsigned short* __restrict__ Uq_bf,
    const float* __restrict__ Up, int t,
    float* __restrict__ r, unsigned short* __restrict__ rbf)
{
  const int b = blockIdx.x;
  __shared__ float sa[LQ_];
  for (int i = threadIdx.x; i < LQ_; i += blockDim.x) sa[i] = a[(long)b * LQ_ + i];
  __syncthreads();

  const unsigned short* U = Uq_bf + (long)b * LQ_ * IN_;
  const int c = threadIdx.x;            // chunk of 8 features, 96*8 = 768
  float acc[8] = {0.f, 0.f, 0.f, 0.f, 0.f, 0.f, 0.f, 0.f};
  for (int l = 0; l < LQ_; ++l) {
    const float w = sa[l];
    union { uint4 u; unsigned short h[8]; } p;
    p.u = *(const uint4*)(U + (long)l * IN_ + c * 8);   // 512B contiguous per wave
    #pragma unroll
    for (int e = 0; e < 8; ++e)
      acc[e] = fmaf(w, bf16_bits_to_f32(p.h[e]), acc[e]);
  }
  #pragma unroll
  for (int e = 0; e < 8; ++e) {
    const long idx = (long)b * (2 * IN_) + IN_ + c * 8 + e;
    r[idx] = acc[e];
    rbf[idx] = f32_to_bf16_bits(acc[e]);
  }
  #pragma unroll
  for (int e = 0; e < 8; ++e) {
    const float u = Up[((long)b * LP_ + t) * IN_ + c * 8 + e];
    const long idx = (long)b * (2 * IN_) + c * 8 + e;
    r[idx] = u;
    rbf[idx] = f32_to_bf16_bits(u);
  }
}

// c_ = sigmoid(glin) * r  (bf16 out, feeds the W_ih GEMM)
__global__ void gate_kernel(const float* __restrict__ glin, const float* __restrict__ r,
                            unsigned short* __restrict__ cgbf, int n) {
  int i = blockIdx.x * blockDim.x + threadIdx.x;
  if (i < n) cgbf[i] = f32_to_bf16_bits(sigmoidf_(glin[i]) * r[i]);
}

// GRU combine; writes v_new (f32 + bf16) and the output slice for step t
__global__ void gru_kernel(const float* __restrict__ gi, const float* __restrict__ gh,
                           const float* __restrict__ v_old,
                           float* __restrict__ v_new, unsigned short* __restrict__ vbf,
                           float* __restrict__ out, int t)
{
  int i = blockIdx.x * blockDim.x + threadIdx.x;
  if (i >= B_ * H_) return;
  int b = i / H_, h = i % H_;
  const float* gib = gi + (long)b * 3 * H_;
  const float* ghb = gh + (long)b * 3 * H_;
  float rg = sigmoidf_(gib[h] + ghb[h]);
  float z  = sigmoidf_(gib[H_ + h] + ghb[H_ + h]);
  float nn = tanhf(gib[2 * H_ + h] + rg * ghb[2 * H_ + h]);
  float vn = (1.0f - z) * nn + z * v_old[i];
  v_new[i] = vn;
  vbf[i] = f32_to_bf16_bits(vn);
  out[((long)b * LP_ + t) * H_ + h] = vn;
}

// ---------- host ----------
extern "C" void kernel_launch(void* const* d_in, const int* in_sizes, int n_in,
                              void* d_out, int out_size, void* d_ws, size_t ws_size,
                              hipStream_t stream)
{
  (void)in_sizes; (void)n_in; (void)out_size; (void)ws_size;
  const float* Up  = (const float*)d_in[0];
  const float* Uq  = (const float*)d_in[1];
  const float* Wp  = (const float*)d_in[3];
  const float* Wq  = (const float*)d_in[4];
  const float* Wv  = (const float*)d_in[5];
  const float* V   = (const float*)d_in[6];
  const float* Wg  = (const float*)d_in[7];
  const float* Wih = (const float*)d_in[8];
  const float* Whh = (const float*)d_in[9];
  const float* bih = (const float*)d_in[10];
  const float* bhh = (const float*)d_in[11];
  float* out = (float*)d_out;

  char* ws = (char*)d_ws;
  size_t off = 0;
  auto alloc = [&](size_t bytes) -> char* {
    char* p = ws + off;
    off = (off + bytes + 255) & ~(size_t)255;
    return p;
  };

  unsigned short* Uq_bf  = (unsigned short*)alloc((size_t)B_ * LQ_ * IN_ * 2);
  unsigned short* Up_bf  = (unsigned short*)alloc((size_t)B_ * LP_ * IN_ * 2);
  unsigned short* WqT    = (unsigned short*)alloc((size_t)H_ * IN_ * 2);
  unsigned short* WpT    = (unsigned short*)alloc((size_t)H_ * IN_ * 2);
  unsigned short* WvT    = (unsigned short*)alloc((size_t)H_ * H_ * 2);
  unsigned short* WgT    = (unsigned short*)alloc((size_t)(2 * IN_) * (2 * IN_) * 2);
  unsigned short* WihT   = (unsigned short*)alloc((size_t)(3 * H_) * (2 * IN_) * 2);
  unsigned short* WhhT   = (unsigned short*)alloc((size_t)(3 * H_) * H_ * 2);
  unsigned short* WuqUqB = (unsigned short*)alloc((size_t)B_ * LQ_ * H_ * 2);
  float*          WupAll = (float*)alloc((size_t)B_ * LP_ * H_ * 4);
  float*          vbufA  = (float*)alloc((size_t)B_ * H_ * 4);
  float*          vbufB  = (float*)alloc((size_t)B_ * H_ * 4);
  unsigned short* vbf    = (unsigned short*)alloc((size_t)B_ * H_ * 2);
  float*          biasBf = (float*)alloc((size_t)B_ * H_ * 4);
  float*          sbuf   = (float*)alloc((size_t)B_ * LQ_ * 4);
  float*          abuf   = (float*)alloc((size_t)B_ * LQ_ * 4);
  float*          rbuf   = (float*)alloc((size_t)B_ * 2 * IN_ * 4);
  unsigned short* rbf    = (unsigned short*)alloc((size_t)B_ * 2 * IN_ * 2);
  float*          glin   = (float*)alloc((size_t)B_ * 2 * IN_ * 4);
  unsigned short* cgbf   = (unsigned short*)alloc((size_t)B_ * 2 * IN_ * 2);
  float*          gibuf  = (float*)alloc((size_t)B_ * 3 * H_ * 4);
  float*          ghbuf  = (float*)alloc((size_t)B_ * 3 * H_ * 4);

  auto launch_gemm = [&](const unsigned short* A, const unsigned short* BT,
                         const float* bias, long bRS, long bOff,
                         float* Cf, unsigned short* Cb, int M, int N, int K) {
    int strips = (M / 16) * (N / 64);
    int blocks = (strips + 3) / 4;     // 4 waves (wave32) per 128-thread block
    wmma_gemm_bf16_kernel<<<blocks, 128, 0, stream>>>(A, BT, bias, bRS, bOff, Cf, Cb, M, N, K);
  };

  // ---- precompute phase (fully parallel) ----
  {
    long n = (long)B_ * LQ_ * IN_;
    convert_to_bf16_kernel<<<(int)((n + 255) / 256), 256, 0, stream>>>(Uq, Uq_bf, n);
    convert_to_bf16_kernel<<<(int)((n + 255) / 256), 256, 0, stream>>>(Up, Up_bf, n);
  }
  auto tr = [&](const float* s, unsigned short* d, int rows, int cols) {
    long n = (long)rows * cols;
    transpose_to_bf16_kernel<<<(int)((n + 255) / 256), 256, 0, stream>>>(s, d, rows, cols);
  };
  tr(Wq,  WqT,  IN_,     H_);
  tr(Wp,  WpT,  IN_,     H_);
  tr(Wv,  WvT,  H_,      H_);
  tr(Wg,  WgT,  2 * IN_, 2 * IN_);
  tr(Wih, WihT, 2 * IN_, 3 * H_);
  tr(Whh, WhhT, H_,      3 * H_);

  // WuqUq = Uq @ Wq (bf16 out; together with Uq_bf ~100 MB -> L2 resident)
  launch_gemm(Uq_bf, WqT, nullptr, 0, 0, nullptr, WuqUqB, B_ * LQ_, H_, IN_);
  // Wup for all t = Up @ Wp (f32, consumed as strided GEMM-epilogue bias)
  launch_gemm(Up_bf, WpT, nullptr, 0, 0, WupAll, nullptr, B_ * LP_, H_, IN_);

  init_v_kernel<<<(B_ * H_ + 255) / 256, 256, 0, stream>>>(vbufA, vbf, B_ * H_);

  // ---- sequential scan over LP steps ----
  float* v_old = vbufA;
  float* v_new = vbufB;
  for (int t = 0; t < LP_; ++t) {
    launch_gemm(vbf, WvT, WupAll, (long)LP_ * H_, (long)t * H_,
                biasBf, nullptr, B_, H_, H_);
    attn_score_kernel<<<B_ * (LQ_ / 8), 256, 0, stream>>>(WuqUqB, biasBf, V, sbuf);
    softmax_kernel<<<B_, LQ_, 0, stream>>>(sbuf, abuf);
    context_kernel<<<B_, 96, 0, stream>>>(abuf, Uq_bf, Up, t, rbuf, rbf);
    launch_gemm(rbf, WgT, nullptr, 0, 0, glin, nullptr, B_, 2 * IN_, 2 * IN_);
    gate_kernel<<<(B_ * 2 * IN_ + 255) / 256, 256, 0, stream>>>(glin, rbuf, cgbf, B_ * 2 * IN_);
    launch_gemm(cgbf, WihT, bih, 0, 0, gibuf, nullptr, B_, 3 * H_, 2 * IN_);
    launch_gemm(vbf,  WhhT, bhh, 0, 0, ghbuf, nullptr, B_, 3 * H_, H_);
    gru_kernel<<<(B_ * H_ + 255) / 256, 256, 0, stream>>>(gibuf, ghbuf, v_old, v_new, vbf, out, t);
    float* tmp = v_old; v_old = v_new; v_new = tmp;
  }
}
